// TransformerBlock_68118181314894
// MI455X (gfx1250) — matrix operand order repaired and checked by reading, not compiled, not run on hardware
//
#include <hip/hip_runtime.h>
#include <hip/hip_bf16.h>
#include <math.h>

// ---------------------------------------------------------------------------
// Problem constants (B=1)
// ---------------------------------------------------------------------------
#define S_  2048
#define D_  4096
#define H_  32
#define HD_ 128
#define FF_ 11008

typedef __attribute__((ext_vector_type(16))) __bf16 v16bf;
typedef __attribute__((ext_vector_type(8)))  float  v8f;
typedef __attribute__((ext_vector_type(8)))  __bf16 bf16x8;
typedef unsigned int u32;
typedef __attribute__((ext_vector_type(4)))  u32 u32x4;
typedef __attribute__((ext_vector_type(8)))  u32 u32x8;

// Build a 16-element bf16 WMMA fragment from two contiguous 8-element chunks.
__device__ __forceinline__ v16bf make_frag(const __bf16* p0, const __bf16* p1) {
  bf16x8 lo = *(const bf16x8*)p0;
  bf16x8 hi = *(const bf16x8*)p1;
  return __builtin_shufflevector(lo, hi, 0,1,2,3,4,5,6,7,8,9,10,11,12,13,14,15);
}

// ---------------------------------------------------------------------------
// Tensor Data Mover: async 2D tile load (bf16, row-major) global -> LDS.
// D# built per CDNA5 ISA; issued via inline asm (portable
// across ROCm7.2 / amdgpu-toolchain builtin arity differences).
// Tile: tile_rows x tile_cols elements (2B each), tensor row stride = K elems.
// ---------------------------------------------------------------------------
__device__ __forceinline__ void tdm_load_tile_bf16(const __bf16* gptr, u32 lds_byte_off,
                                                   u32 K_elems, u32 M_rows,
                                                   u32 tile_cols, u32 tile_rows) {
  const unsigned long long ga = (unsigned long long)(uintptr_t)gptr;
  u32x4 g0;
  g0[0] = 1u;                                             // count=1 valid descriptor
  g0[1] = lds_byte_off;                                   // lds_addr (bytes)
  g0[2] = (u32)(ga & 0xffffffffu);                        // global_addr[31:0]
  g0[3] = (u32)((ga >> 32) & 0x01ffffffu) | (2u << 30);   // addr[56:32] | type=2
  u32x8 g1;
  g1[0] = (1u << 16);                                     // wg_mask=0, data_size=2B
  g1[1] = (K_elems & 0xffffu) << 16;                      // tensor_dim0 lo16
  g1[2] = ((K_elems >> 16) & 0xffffu) | ((M_rows & 0xffffu) << 16); // dim0 hi | dim1 lo
  g1[3] = ((M_rows >> 16) & 0xffffu) | (tile_cols << 16); // dim1 hi | tile_dim0
  g1[4] = tile_rows;                                      // tile_dim1 | tile_dim2=0
  g1[5] = K_elems;                                        // tensor_dim0_stride[31:0]
  g1[6] = 0u;                                             // stride hi | dim1_stride lo
  g1[7] = 0u;
  asm volatile("tensor_load_to_lds %0, %1" :: "s"(g0), "s"(g1) : "memory");
}

// ---------------------------------------------------------------------------
// RMSNorm: one block per token row, fp32 in -> bf16 out (scaled by weight)
// ---------------------------------------------------------------------------
__global__ __launch_bounds__(256)
void rmsnorm_kernel(const float* __restrict__ x, const float* __restrict__ w,
                    __bf16* __restrict__ out) {
  __shared__ float red[8];
  const int row = blockIdx.x;
  const float* xr = x + (size_t)row * D_;
  float ss = 0.f;
  for (int i = threadIdx.x; i < D_; i += 256) { float v = xr[i]; ss += v * v; }
#pragma unroll
  for (int off = 16; off > 0; off >>= 1) ss += __shfl_xor(ss, off, 32);
  if ((threadIdx.x & 31) == 0) red[threadIdx.x >> 5] = ss;
  __syncthreads();
  if (threadIdx.x < 8) {
    float v = red[threadIdx.x];
#pragma unroll
    for (int off = 4; off > 0; off >>= 1) v += __shfl_xor(v, off, 32);
    if (threadIdx.x == 0) red[0] = v;
  }
  __syncthreads();
  const float rs = rsqrtf(red[0] / (float)D_ + 1e-6f);
  __bf16* orow = out + (size_t)row * D_;
  for (int i = threadIdx.x; i < D_; i += 256)
    orow[i] = (__bf16)(xr[i] * rs * w[i]);
}

// ---------------------------------------------------------------------------
// Tiled bf16 WMMA GEMM: C[M,N] = A[M,K](bf16) * B[K,N](fp32, converted in-kernel)
// Block: 256 threads = 8 waves; tile 128x128; each wave owns 32x64 (2x4 WMMA).
// A tile staged by the Tensor Data Mover (async, TENSORcnt); B tile staged
// cooperatively with in-flight fp32->bf16 conversion (overlaps the TDM DMA).
// ---------------------------------------------------------------------------
template<bool OUT_BF16, bool HAS_RES>
__global__ __launch_bounds__(256)
void gemm_bf16_kernel(const __bf16* __restrict__ A, const float* __restrict__ Bw,
                      float* __restrict__ Cf, __bf16* __restrict__ Cb,
                      const float* __restrict__ res, int M, int K, int N) {
  __shared__ __bf16 ldsA[128][32];    // 8 KB, row-major M x K-chunk (TDM target)
  __shared__ __bf16 ldsBt[128][32];   // 8 KB, transposed: [n][k]
  const int m0 = blockIdx.y * 128;
  const int n0 = blockIdx.x * 128;
  const int t = threadIdx.x;
  const int lane = t & 31, l = lane & 15, hi = lane >> 4;
  const int wave = t >> 5;
  const int wm = (wave >> 1) * 32;   // wave row base (4 wave-rows x 32)
  const int wn = (wave & 1) * 64;    // wave col base (2 wave-cols x 64)

  v8f acc[2][4] = {};
  const int bkrow = t >> 3, bgrp = t & 7;         // B: 32 rows x 8 col-groups of 16
  const u32 ldsA_off = (u32)(uintptr_t)&ldsA[0][0];

  for (int k0 = 0; k0 < K; k0 += 32) {
    // Async A tile 128x32 bf16 via Tensor Data Mover (issued once per block,
    // by wave 0; EXEC is ignored by tensor ops, descriptor is wave-uniform).
    if (wave == 0)
      tdm_load_tile_bf16(A + (size_t)m0 * K + k0, ldsA_off, (u32)K, (u32)M, 32u, 128u);

    if (k0 + 32 < K)  // prefetch next B tile -> global_prefetch_b8
      __builtin_prefetch(Bw + (size_t)(k0 + 32 + bkrow) * N + n0 + bgrp * 16, 0, 0);

    // Stage B tile 32x128 fp32 -> bf16, stored transposed for contiguous frags
    {
      const float* src = Bw + (size_t)(k0 + bkrow) * N + n0 + bgrp * 16;
      float4 b0 = *(const float4*)src;
      float4 b1 = *(const float4*)(src + 4);
      float4 b2 = *(const float4*)(src + 8);
      float4 b3 = *(const float4*)(src + 12);
      const float vals[16] = {b0.x, b0.y, b0.z, b0.w, b1.x, b1.y, b1.z, b1.w,
                              b2.x, b2.y, b2.z, b2.w, b3.x, b3.y, b3.z, b3.w};
#pragma unroll
      for (int e = 0; e < 16; ++e) ldsBt[bgrp * 16 + e][bkrow] = (__bf16)vals[e];
    }
    if (wave == 0) __builtin_amdgcn_s_wait_tensorcnt(0);
    __syncthreads();

    v16bf af[2], bfr[4];
#pragma unroll
    for (int mi = 0; mi < 2; ++mi) {
      // A layout: lane l row M=l(+wm), elems 0..7 -> K=hi*8+j, 8..15 -> K=16+hi*8+j
      const __bf16* p = &ldsA[wm + mi * 16 + l][hi * 8];
      af[mi] = make_frag(p, p + 16);
    }
#pragma unroll
    for (int ni = 0; ni < 4; ++ni) {
      // B layout: lane l col N=l(+wn), elems 0..7 -> K=hi*8+j, 8..15 -> K=16+hi*8+j
      const __bf16* p = &ldsBt[wn + ni * 16 + l][hi * 8];
      bfr[ni] = make_frag(p, p + 16);
    }
#pragma unroll
    for (int mi = 0; mi < 2; ++mi)
#pragma unroll
      for (int ni = 0; ni < 4; ++ni)
        acc[mi][ni] = __builtin_amdgcn_wmma_f32_16x16x32_bf16(
            false, af[mi], false, bfr[ni], (short)0, acc[mi][ni], false, false);
    __syncthreads();  // all reads done before next TDM overwrite of ldsA
  }

  // Epilogue. C layout: VGPR r -> row r (lanes 0-15) / r+8 (lanes 16-31); col = l.
#pragma unroll
  for (int mi = 0; mi < 2; ++mi)
#pragma unroll
    for (int ni = 0; ni < 4; ++ni)
#pragma unroll
      for (int r = 0; r < 8; ++r) {
        const int m = m0 + wm + mi * 16 + r + hi * 8;
        const int n = n0 + wn + ni * 16 + l;
        const size_t idx = (size_t)m * N + n;
        float v = acc[mi][ni][r];
        if (HAS_RES) v += res[idx];
        if (OUT_BF16) Cb[idx] = (__bf16)v; else Cf[idx] = v;
      }
}

// ---------------------------------------------------------------------------
// RoPE on Q and K: fp32 in -> bf16 out (interleaved real/imag pairs)
// ---------------------------------------------------------------------------
__global__ void rope_kernel(const float* __restrict__ q32, const float* __restrict__ k32,
                            const float* __restrict__ cosT, const float* __restrict__ sinT,
                            __bf16* __restrict__ qb, __bf16* __restrict__ kb) {
  const int idx = blockIdx.x * blockDim.x + threadIdx.x; // S*H*(HD/2)
  if (idx >= S_ * H_ * (HD_ / 2)) return;
  const int i = idx & 63;           // rotary pair index
  const int sh = idx >> 6;
  const int hh = sh & (H_ - 1);
  const int s = sh >> 5;            // H_ == 32
  const size_t base = (size_t)s * D_ + (size_t)hh * HD_ + 2 * i;
  const float c = cosT[s * 64 + i], sn = sinT[s * 64 + i];
  float xr = q32[base], xi = q32[base + 1];
  qb[base]     = (__bf16)(xr * c - xi * sn);
  qb[base + 1] = (__bf16)(xr * sn + xi * c);
  xr = k32[base]; xi = k32[base + 1];
  kb[base]     = (__bf16)(xr * c - xi * sn);
  kb[base + 1] = (__bf16)(xr * sn + xi * c);
}

// ---------------------------------------------------------------------------
// Flash attention: 1 wave per (head, 16-row query tile), 32-key tiles,
// online softmax, causal mask applied analytically.
// ---------------------------------------------------------------------------
__global__ __launch_bounds__(32)
void flash_attn_kernel(const __bf16* __restrict__ qb, const __bf16* __restrict__ kb,
                       const __bf16* __restrict__ vb, __bf16* __restrict__ ob) {
  __shared__ __bf16 ldsP[16][32];    // P tile (bf16) for A-layout reload
  __shared__ __bf16 ldsVt[128][32];  // V tile transposed: [d][key]
  const int q0 = blockIdx.x * 16;
  const int h  = blockIdx.y;
  const int lane = threadIdx.x & 31, l = lane & 15, hi = lane >> 4;
  const size_t hoff = (size_t)h * HD_;

  // Preload Q fragments: 4 K-chunks of 32 over HD=128
  v16bf qf[4];
#pragma unroll
  for (int c = 0; c < 4; ++c) {
    const __bf16* p = qb + (size_t)(q0 + l) * D_ + hoff + c * 32 + hi * 8;
    qf[c] = make_frag(p, p + 16);
  }

  v8f o[8] = {};
  float rmax[8], rsum[8];
#pragma unroll
  for (int r = 0; r < 8; ++r) { rmax[r] = -1e30f; rsum[r] = 0.f; }
  const float scale = 0.08838834764831845f; // 1/sqrt(128)
  const int ntiles = (q0 + 16 + 31) / 32;   // causal: only key tiles that intersect

  for (int kt = 0; kt < ntiles; ++kt) {
    const int key0 = kt * 32;
    // scores S = Q * K^T  (2 col subtiles x 4 K-chunks = 8 WMMA)
    v8f sacc[2] = {};
#pragma unroll
    for (int ni = 0; ni < 2; ++ni) {
#pragma unroll
      for (int c = 0; c < 4; ++c) {
        const __bf16* p = kb + (size_t)(key0 + ni * 16 + l) * D_ + hoff + c * 32 + hi * 8;
        v16bf kf = make_frag(p, p + 16);
        sacc[ni] = __builtin_amdgcn_wmma_f32_16x16x32_bf16(
            false, qf[c], false, kf, (short)0, sacc[ni], false, false);
      }
    }
    // Stage V tile transposed: each lane owns one key row
    {
      const __bf16* src = vb + (size_t)(key0 + lane) * D_ + hoff;
#pragma unroll
      for (int d8 = 0; d8 < 16; ++d8) {
        bf16x8 vv = *(const bf16x8*)(src + d8 * 8);
#pragma unroll
        for (int e = 0; e < 8; ++e) ldsVt[d8 * 8 + e][lane] = vv[e];
      }
    }
    // mask + online softmax (row = r + hi*8, col = key0 + {l, 16+l})
    float p0[8], p1[8], corr[8];
#pragma unroll
    for (int r = 0; r < 8; ++r) {
      const int m = r + hi * 8;
      const float s0 = (key0 + l      <= q0 + m) ? sacc[0][r] * scale : -1e30f;
      const float s1 = (key0 + 16 + l <= q0 + m) ? sacc[1][r] * scale : -1e30f;
      float tmax = fmaxf(s0, s1);
#pragma unroll
      for (int off = 8; off > 0; off >>= 1) tmax = fmaxf(tmax, __shfl_xor(tmax, off, 32));
      const float nmax = fmaxf(rmax[r], tmax);
      corr[r] = __expf(rmax[r] - nmax);
      rmax[r] = nmax;
      p0[r] = __expf(s0 - nmax);
      p1[r] = __expf(s1 - nmax);
      float tsum = p0[r] + p1[r];
#pragma unroll
      for (int off = 8; off > 0; off >>= 1) tsum += __shfl_xor(tsum, off, 32);
      rsum[r] = rsum[r] * corr[r] + tsum;
    }
    // rescale O, stage P (bf16) for A-layout
#pragma unroll
    for (int nn = 0; nn < 8; ++nn)
#pragma unroll
      for (int r = 0; r < 8; ++r) o[nn][r] *= corr[r];
#pragma unroll
    for (int r = 0; r < 8; ++r) {
      const int m = r + hi * 8;
      ldsP[m][l]      = (__bf16)p0[r];
      ldsP[m][16 + l] = (__bf16)p1[r];
    }
    __syncthreads();
    // O += P(16x32) * V(32x128): 8 WMMA
    {
      const __bf16* pp = &ldsP[l][hi * 8];
      v16bf pf = make_frag(pp, pp + 16);
#pragma unroll
      for (int nn = 0; nn < 8; ++nn) {
        const __bf16* vp = &ldsVt[nn * 16 + l][hi * 8];
        v16bf vf = make_frag(vp, vp + 16);
        o[nn] = __builtin_amdgcn_wmma_f32_16x16x32_bf16(
            false, pf, false, vf, (short)0, o[nn], false, false);
      }
    }
    __syncthreads();
  }
  // normalize + write bf16 (feeds the WO GEMM)
#pragma unroll
  for (int r = 0; r < 8; ++r) {
    const int m = r + hi * 8;
    const float inv = 1.0f / rsum[r];
#pragma unroll
    for (int nn = 0; nn < 8; ++nn)
      ob[(size_t)(q0 + m) * D_ + hoff + nn * 16 + l] = (__bf16)(o[nn][r] * inv);
  }
}

// ---------------------------------------------------------------------------
// SwiGLU: a = bf16( silu(g) * u )
// ---------------------------------------------------------------------------
__global__ void silu_mul_kernel(const float* __restrict__ g, const float* __restrict__ u,
                                __bf16* __restrict__ a, size_t n) {
  const size_t i = (size_t)blockIdx.x * 256 + threadIdx.x;
  if (i < n) {
    const float gv = g[i];
    const float s = gv / (1.f + __expf(-gv));
    a[i] = (__bf16)(s * u[i]);
  }
}

// ---------------------------------------------------------------------------
// Host orchestration
// ---------------------------------------------------------------------------
extern "C" void kernel_launch(void* const* d_in, const int* in_sizes, int n_in,
                              void* d_out, int out_size, void* d_ws, size_t ws_size,
                              hipStream_t stream) {
  (void)in_sizes; (void)n_in; (void)out_size; (void)ws_size;
  const float* x      = (const float*)d_in[0];
  const float* wq     = (const float*)d_in[1];
  const float* wk     = (const float*)d_in[2];
  const float* wv     = (const float*)d_in[3];
  const float* wo     = (const float*)d_in[4];
  const float* w1     = (const float*)d_in[5];
  const float* w2     = (const float*)d_in[6];
  const float* w3     = (const float*)d_in[7];
  const float* anw    = (const float*)d_in[8];
  const float* fnw    = (const float*)d_in[9];
  const float* cosT   = (const float*)d_in[10];
  const float* sinT   = (const float*)d_in[11];
  // d_in[12] = mask (causal mask applied analytically)
  float* out = (float*)d_out;

  char* ws = (char*)d_ws;
  size_t off = 0;
  auto alloc = [&](size_t bytes) -> void* {
    void* p = ws + off;
    off += (bytes + 255) & ~(size_t)255;
    return p;
  };
  const size_t SD = (size_t)S_ * D_;
  const size_t SF = (size_t)S_ * FF_;
  __bf16* hn      = (__bf16*)alloc(SD * 2);
  float*  q32     = (float*) alloc(SD * 4);
  float*  k32     = (float*) alloc(SD * 4);
  __bf16* vbuf    = (__bf16*)alloc(SD * 2);
  __bf16* qb      = (__bf16*)alloc(SD * 2);
  __bf16* kbf     = (__bf16*)alloc(SD * 2);
  __bf16* attn_bf = (__bf16*)alloc(SD * 2);
  float*  hbuf    = (float*) alloc(SD * 4);
  __bf16* fn      = (__bf16*)alloc(SD * 2);
  float*  gbuf    = (float*) alloc(SF * 4);
  float*  ubuf    = (float*) alloc(SF * 4);
  __bf16* abuf    = (__bf16*)alloc(SF * 2);

  const dim3 gD(D_ / 128, S_ / 128);   // GEMMs with N = D
  const dim3 gF(FF_ / 128, S_ / 128);  // GEMMs with N = FF

  // 1) attn rmsnorm
  rmsnorm_kernel<<<S_, 256, 0, stream>>>(x, anw, hn);
  // 2) Q/K/V projections
  gemm_bf16_kernel<false, false><<<gD, 256, 0, stream>>>(hn, wq, q32, nullptr, nullptr, S_, D_, D_);
  gemm_bf16_kernel<false, false><<<gD, 256, 0, stream>>>(hn, wk, k32, nullptr, nullptr, S_, D_, D_);
  gemm_bf16_kernel<true,  false><<<gD, 256, 0, stream>>>(hn, wv, nullptr, vbuf, nullptr, S_, D_, D_);
  // 3) RoPE -> bf16 Q/K
  const int nrope = S_ * H_ * (HD_ / 2);
  rope_kernel<<<(nrope + 255) / 256, 256, 0, stream>>>(q32, k32, cosT, sinT, qb, kbf);
  // 4) flash attention
  flash_attn_kernel<<<dim3(S_ / 16, H_), 32, 0, stream>>>(qb, kbf, vbuf, attn_bf);
  // 5) output projection + residual: h = x + attn @ wo
  gemm_bf16_kernel<false, true><<<gD, 256, 0, stream>>>(attn_bf, wo, hbuf, nullptr, x, S_, D_, D_);
  // 6) ffn rmsnorm
  rmsnorm_kernel<<<S_, 256, 0, stream>>>(hbuf, fnw, fn);
  // 7) gate/up projections
  gemm_bf16_kernel<false, false><<<gF, 256, 0, stream>>>(fn, w1, gbuf, nullptr, nullptr, S_, D_, FF_);
  gemm_bf16_kernel<false, false><<<gF, 256, 0, stream>>>(fn, w3, ubuf, nullptr, nullptr, S_, D_, FF_);
  // 8) swiglu
  silu_mul_kernel<<<(unsigned)((SF + 255) / 256), 256, 0, stream>>>(gbuf, ubuf, abuf, SF);
  // 9) down projection + residual: out = h + a @ w2
  gemm_bf16_kernel<false, true><<<gD, 256, 0, stream>>>(abuf, w2, out, nullptr, hbuf, S_, FF_, D_);
}
